// BFR3_28089086116617
// MI455X (gfx1250) — matrix operand order repaired and checked by reading, not compiled, not run on hardware
//
#include <hip/hip_runtime.h>
#include <hip/hip_bf16.h>
#include <cstddef>

#define Bsz   4
#define Gn    1200
#define NIN   10
#define Hd    4
#define NHE   300
#define NINC  4800
#define ALPHA 0.005f
#define BETA  5e-5f
#define BN_EPS 1e-5f
#define NTILE (Gn / 16)        // 75 row tiles
#define KFULL ((Gn / 32) * 32) // 1184 : start of the partial K-chunk
#define NCHUNK ((Gn + 31) / 32) // 38 K-chunks of 32

typedef __attribute__((ext_vector_type(16))) _Float16 v16h;
typedef __attribute__((ext_vector_type(2)))  __fp16   v2hf;  // cvt_pkrtz result type
typedef __attribute__((ext_vector_type(8)))  float    v8f;

union AFrag { v16h v; v2hf p[8]; };

__device__ __forceinline__ float elu1(float v) {
    return v > 0.f ? v : (__expf(v) - 1.f);
}

// fast sigmoid: v_exp_f32 + v_rcp_f32 (no IEEE div sequence)
__device__ __forceinline__ float fast_sigmoid(float x) {
    return __builtin_amdgcn_rcpf(1.f + __expf(-x));
}

__device__ __forceinline__ float edge_mask(float ed, float coef) {
    return (ed == 0.f) ? coef : ed;   // edge + coef*(edge==0), edge in {0,1}
}

// ---------------------------------------------------------------------------
// K1: h = elu(x @ infer_w^T + infer_b); gate projections
//     a[j] = ew[0:4]·h_j ; b[i] = ew[4:8]·h_i + eb
// ---------------------------------------------------------------------------
__global__ void prep1_kernel(const float* __restrict__ x,
                             const float* __restrict__ iw, const float* __restrict__ ib,
                             const float* __restrict__ ew, const float* __restrict__ eb,
                             float* __restrict__ h, float* __restrict__ av,
                             float* __restrict__ bv) {
    int idx = blockIdx.x * blockDim.x + threadIdx.x;   // b*G+g
    if (idx >= Bsz * Gn) return;
    float hv[Hd];
    #pragma unroll
    for (int o = 0; o < Hd; ++o) {
        float acc = ib[o];
        #pragma unroll
        for (int n = 0; n < NIN; ++n) acc += iw[o * NIN + n] * x[(size_t)idx * NIN + n];
        hv[o] = elu1(acc);
        h[(size_t)idx * Hd + o] = hv[o];
    }
    float a = 0.f, bb = eb[0];
    #pragma unroll
    for (int o = 0; o < Hd; ++o) { a += ew[o] * hv[o]; bb += ew[Hd + o] * hv[o]; }
    av[idx] = a;
    bv[idx] = bb;
}

// ---------------------------------------------------------------------------
// B-fragment prepack: for each (batch, K-chunk, lane) build the per-lane v16h
// slice of X[32x16] (cols 0..3 = h_j, col 4 = 1.0, rest 0; j>=G zeroed).
// Layout matches WMMA 16-bit B operand: N = lane&15, K = (lane>>4)*8 + (e&7) + (e>>3)*16.
// ---------------------------------------------------------------------------
__global__ void bprep_kernel(const float* __restrict__ h, v16h* __restrict__ bfrag) {
    int t = blockIdx.x * blockDim.x + threadIdx.x;     // b*NCHUNK*32 + chunk*32 + lane
    if (t >= Bsz * NCHUNK * 32) return;
    const int lane  = t & 31;
    const int chunk = (t >> 5) % NCHUNK;
    const int b     = t / (NCHUNK * 32);
    const int n     = lane & 15;
    const int khalf = lane >> 4;
    const int jb2   = chunk * 32 + khalf * 8;
    v16h v;
    #pragma unroll
    for (int e = 0; e < 16; ++e) {
        const int j = jb2 + (e & 7) + (e >> 3) * 16;
        float val = 0.f;
        if (j < Gn) {
            if (n < Hd)       val = h[((size_t)b * Gn + j) * Hd + n];
            else if (n == Hd) val = 1.f;
        }
        v[e] = (_Float16)val;
    }
    bfrag[t] = v;
}

// ---------------------------------------------------------------------------
// Heavy round: one wave per 16-row target tile.
//   A tile (16x32 f16): S[i][j] = sigmoid(a_j + b_i) * mask_ij (fp32 gate math)
//   B tile (32x16 f16): preloaded fragments (X | ones column)
//   C (16x16 f32) via v_wmma_f32_16x16x32_f16 over K = G.
//   recv = [C[:,0:4], x_i * C[:,4]] -> nodes(8->4,elu) -> mm(cat(t,x_i))(8->4,elu)
// ---------------------------------------------------------------------------
__global__ __launch_bounds__(32) void round_kernel(
        const float* __restrict__ h,    // [B,G,H]
        const float* __restrict__ av,   // [B,G]
        const float* __restrict__ bv,   // [B,G]
        const float* __restrict__ edge, // [G,G]
        const v16h*  __restrict__ bfrag,// [B,NCHUNK,32]
        float coef,
        const float* __restrict__ nw, const float* __restrict__ nb,  // [H,2H],[H]
        const float* __restrict__ mw, const float* __restrict__ mb,  // [H,2H],[H]
        float* __restrict__ upd)        // [B,G,H]
{
    const int blk  = blockIdx.x;        // 0 .. B*NTILE-1
    const int b    = blk / NTILE;
    const int tile = blk % NTILE;
    const int i0   = tile * 16;
    const int lane = threadIdx.x;       // 0..31
    const int m     = lane & 15;        // A: row; C: column n
    const int khalf = lane >> 4;
    const int i     = i0 + m;
    const float bi  = bv[(size_t)b * Gn + i];

    const float* __restrict__ erow = edge + (size_t)i * Gn;
    const float* __restrict__ avb  = av + (size_t)b * Gn;
    const v16h*  __restrict__ bfb  = bfrag + (size_t)b * NCHUNK * 32 + lane;

    v8f c = {};

    // ---- 37 full K-chunks: no guards, no exec manipulation ---------------
    #pragma unroll 2
    for (int k0 = 0; k0 < KFULL; k0 += 32) {
        AFrag a;
        #pragma unroll
        for (int grp = 0; grp < 2; ++grp) {
            const int jb = k0 + grp * 16 + khalf * 8;
            const float4* ep = reinterpret_cast<const float4*>(erow + jb);
            const float4* ap = reinterpret_cast<const float4*>(avb + jb);
            float4 e0 = ep[0], e1 = ep[1];
            float4 a0 = ap[0], a1 = ap[1];
            float ed[8] = {e0.x, e0.y, e0.z, e0.w, e1.x, e1.y, e1.z, e1.w};
            float aj[8] = {a0.x, a0.y, a0.z, a0.w, a1.x, a1.y, a1.z, a1.w};
            #pragma unroll
            for (int q = 0; q < 8; q += 2) {
                float s0 = fast_sigmoid(aj[q]     + bi) * edge_mask(ed[q],     coef);
                float s1 = fast_sigmoid(aj[q + 1] + bi) * edge_mask(ed[q + 1], coef);
                a.p[grp * 4 + q / 2] = __builtin_amdgcn_cvt_pkrtz(s0, s1);
            }
        }
        v16h bf = bfb[(k0 >> 5) * 32];
        c = __builtin_amdgcn_wmma_f32_16x16x32_f16(
                false, a.v, false, bf, (short)0, c, false, false);
    }

    // ---- tail chunk k0 = KFULL: only grp 0 (j = KFULL + khalf*8 + q) valid.
    //      Validity depends only on grp -> compile-time, still no divergence.
    {
        AFrag a;
        const int jb = KFULL + khalf * 8;        // max 1192; +7 = 1199 < G
        const float4* ep = reinterpret_cast<const float4*>(erow + jb);
        const float4* ap = reinterpret_cast<const float4*>(avb + jb);
        float4 e0 = ep[0], e1 = ep[1];
        float4 a0 = ap[0], a1 = ap[1];
        float ed[8] = {e0.x, e0.y, e0.z, e0.w, e1.x, e1.y, e1.z, e1.w};
        float aj[8] = {a0.x, a0.y, a0.z, a0.w, a1.x, a1.y, a1.z, a1.w};
        #pragma unroll
        for (int q = 0; q < 8; q += 2) {
            float s0 = fast_sigmoid(aj[q]     + bi) * edge_mask(ed[q],     coef);
            float s1 = fast_sigmoid(aj[q + 1] + bi) * edge_mask(ed[q + 1], coef);
            a.p[q / 2] = __builtin_amdgcn_cvt_pkrtz(s0, s1);
        }
        #pragma unroll
        for (int q = 4; q < 8; ++q) {
            a.p[q] = __builtin_amdgcn_cvt_pkrtz(0.f, 0.f);   // grp 1 -> j >= G, zero
        }
        v16h bf = bfb[(KFULL >> 5) * 32];          // prep already zeroed j >= G
        c = __builtin_amdgcn_wmma_f32_16x16x32_f16(
                false, a.v, false, bf, (short)0, c, false, false);
    }

    // ---- C tile -> LDS -> per-row epilogue --------------------------------
    __shared__ float Csh[16][17];
    #pragma unroll
    for (int r = 0; r < 8; ++r) Csh[khalf * 8 + r][m] = c[r];
    __syncthreads();

    if (lane < 16) {
        const int ii = i0 + lane;
        const float rs = Csh[lane][Hd];       // row-sum column (ones trick)
        float xi[Hd], recv[2 * Hd];
        #pragma unroll
        for (int hh = 0; hh < Hd; ++hh) {
            xi[hh]        = h[((size_t)b * Gn + ii) * Hd + hh];
            recv[hh]      = Csh[lane][hh];    // sum_j s_ij * x_j
            recv[Hd + hh] = xi[hh] * rs;      // x_i * sum_j s_ij
        }
        float t[Hd];
        #pragma unroll
        for (int o = 0; o < Hd; ++o) {
            float acc = nb[o];
            #pragma unroll
            for (int q = 0; q < 2 * Hd; ++q) acc += nw[o * 2 * Hd + q] * recv[q];
            t[o] = elu1(acc);
        }
        #pragma unroll
        for (int o = 0; o < Hd; ++o) {
            float acc = mb[o];
            #pragma unroll
            for (int q = 0; q < Hd; ++q)
                acc += mw[o * 2 * Hd + q] * t[q] + mw[o * 2 * Hd + Hd + q] * xi[q];
            upd[((size_t)b * Gn + ii) * Hd + o] = elu1(acc);
        }
    }
}

// ---------------------------------------------------------------------------
// K3: h2 = bn(elu(upd1 * diag(W1)[g] + b1[g])) per gene over (batch,feat);
//     plus round-2 gate projections.
// ---------------------------------------------------------------------------
__global__ void mid_kernel(const float* __restrict__ upd1,
                           const float* __restrict__ W1, const float* __restrict__ b1,
                           const float* __restrict__ ew2, const float* __restrict__ eb2,
                           float* __restrict__ h2, float* __restrict__ av2,
                           float* __restrict__ bv2) {
    int g = blockIdx.x * blockDim.x + threadIdx.x;
    if (g >= Gn) return;
    const float w1d = W1[(size_t)g * Gn + g];
    const float bb  = b1[g];
    float vals[Bsz * Hd];
    float mean = 0.f;
    #pragma unroll
    for (int b = 0; b < Bsz; ++b)
        #pragma unroll
        for (int hh = 0; hh < Hd; ++hh) {
            float v = elu1(upd1[((size_t)b * Gn + g) * Hd + hh] * w1d + bb);
            vals[b * Hd + hh] = v;
            mean += v;
        }
    mean *= (1.f / (Bsz * Hd));
    float var = 0.f;
    #pragma unroll
    for (int t = 0; t < Bsz * Hd; ++t) { float d = vals[t] - mean; var += d * d; }
    var *= (1.f / (Bsz * Hd));
    const float inv = rsqrtf(var + BN_EPS);
    #pragma unroll
    for (int b = 0; b < Bsz; ++b) {
        float a = 0.f, bbv = eb2[0];
        #pragma unroll
        for (int hh = 0; hh < Hd; ++hh) {
            float vn = (vals[b * Hd + hh] - mean) * inv;
            h2[((size_t)b * Gn + g) * Hd + hh] = vn;
            a   += ew2[hh] * vn;
            bbv += ew2[Hd + hh] * vn;
        }
        av2[(size_t)b * Gn + g] = a;
        bv2[(size_t)b * Gn + g] = bbv;
    }
}

// ---------------------------------------------------------------------------
// K5: upd2 = bn(upd2raw) per gene; xf = upd2 @ hg_w^T
// ---------------------------------------------------------------------------
__global__ void bn2_kernel(const float* __restrict__ raw,
                           const float* __restrict__ hgw,
                           float* __restrict__ upd2, float* __restrict__ xf) {
    int g = blockIdx.x * blockDim.x + threadIdx.x;
    if (g >= Gn) return;
    float vals[Bsz * Hd];
    float mean = 0.f;
    #pragma unroll
    for (int b = 0; b < Bsz; ++b)
        #pragma unroll
        for (int hh = 0; hh < Hd; ++hh) {
            float v = raw[((size_t)b * Gn + g) * Hd + hh];
            vals[b * Hd + hh] = v;
            mean += v;
        }
    mean *= (1.f / (Bsz * Hd));
    float var = 0.f;
    #pragma unroll
    for (int t = 0; t < Bsz * Hd; ++t) { float d = vals[t] - mean; var += d * d; }
    var *= (1.f / (Bsz * Hd));
    const float inv = rsqrtf(var + BN_EPS);
    #pragma unroll
    for (int b = 0; b < Bsz; ++b) {
        float vn[Hd];
        #pragma unroll
        for (int hh = 0; hh < Hd; ++hh) {
            vn[hh] = (vals[b * Hd + hh] - mean) * inv;
            upd2[((size_t)b * Gn + g) * Hd + hh] = vn[hh];
        }
        #pragma unroll
        for (int cc = 0; cc < Hd; ++cc) {
            float acc = 0.f;
            #pragma unroll
            for (int hh = 0; hh < Hd; ++hh) acc += vn[hh] * hgw[cc * Hd + hh];
            xf[((size_t)b * Gn + g) * Hd + cc] = acc;
        }
    }
}

// ---------------------------------------------------------------------------
// K6: efeat[e] = binv(e) * sum over incidences (deterministic scan)
// ---------------------------------------------------------------------------
__global__ void hyperedge_kernel(const float* __restrict__ xf,
                                 const int* __restrict__ hn, const int* __restrict__ he,
                                 float* __restrict__ efeat) {
    int e = blockIdx.x * blockDim.x + threadIdx.x;
    if (e >= NHE) return;
    float s[Hd] = {0.f, 0.f, 0.f, 0.f};
    int cnt = 0;
    for (int k = 0; k < NINC; ++k) {
        if (he[k] == e) {
            ++cnt;
            const int gg = hn[k];
            #pragma unroll
            for (int b = 0; b < Bsz; ++b) {
                const float* p = xf + ((size_t)b * Gn + gg) * Hd;
                #pragma unroll
                for (int cc = 0; cc < Hd; ++cc) s[cc] += p[cc];
            }
        }
    }
    const float binv = (cnt > 0) ? 1.f / ((float)cnt * (float)Bsz) : 0.f;
    #pragma unroll
    for (int cc = 0; cc < Hd; ++cc) efeat[(size_t)e * Hd + cc] = s[cc] * binv;
}

// ---------------------------------------------------------------------------
// K7: hx = elu(dinv * gather(efeat) + hg_b); out = elu(cat(upd2,hx) @ mm3^T + b)
// ---------------------------------------------------------------------------
__global__ void final_kernel(const float* __restrict__ upd2,
                             const float* __restrict__ efeat,
                             const int* __restrict__ hn, const int* __restrict__ he,
                             const float* __restrict__ hgb,
                             const float* __restrict__ m3w, const float* __restrict__ m3b,
                             float* __restrict__ out) {
    int idx = blockIdx.x * blockDim.x + threadIdx.x;  // b*G+g
    if (idx >= Bsz * Gn) return;
    const int g = idx % Gn;
    float s[Hd] = {0.f, 0.f, 0.f, 0.f};
    int cnt = 0;
    for (int k = 0; k < NINC; ++k) {
        if (hn[k] == g) {
            ++cnt;
            const float* p = efeat + (size_t)he[k] * Hd;
            #pragma unroll
            for (int cc = 0; cc < Hd; ++cc) s[cc] += p[cc];
        }
    }
    const float dinv = (cnt > 0) ? 1.f / (float)cnt : 0.f;
    float hx[Hd], u[Hd];
    #pragma unroll
    for (int cc = 0; cc < Hd; ++cc) {
        hx[cc] = elu1(s[cc] * dinv + hgb[cc]);
        u[cc]  = upd2[(size_t)idx * Hd + cc];
    }
    #pragma unroll
    for (int o = 0; o < Hd; ++o) {
        float acc = m3b[o];
        #pragma unroll
        for (int cc = 0; cc < Hd; ++cc)
            acc += m3w[o * 2 * Hd + cc] * u[cc] + m3w[o * 2 * Hd + Hd + cc] * hx[cc];
        out[(size_t)idx * Hd + o] = elu1(acc);
    }
}

// ---------------------------------------------------------------------------
extern "C" void kernel_launch(void* const* d_in, const int* in_sizes, int n_in,
                              void* d_out, int out_size, void* d_ws, size_t ws_size,
                              hipStream_t stream) {
    const float* x        = (const float*)d_in[0];
    const float* edge1    = (const float*)d_in[1];
    const float* edge2    = (const float*)d_in[2];
    const float* W1       = (const float*)d_in[3];
    const float* b1       = (const float*)d_in[4];
    const float* infer_w  = (const float*)d_in[5];
    const float* infer_b  = (const float*)d_in[6];
    const float* mlp_e1_w = (const float*)d_in[7];
    const float* mlp_e1_b = (const float*)d_in[8];
    const float* mlp_e2_w = (const float*)d_in[9];
    const float* mlp_e2_b = (const float*)d_in[10];
    const float* nodes1_w = (const float*)d_in[11];
    const float* nodes1_b = (const float*)d_in[12];
    const float* nodes2_w = (const float*)d_in[13];
    const float* nodes2_b = (const float*)d_in[14];
    const float* mm1_w    = (const float*)d_in[15];
    const float* mm1_b    = (const float*)d_in[16];
    const float* mm2_w    = (const float*)d_in[17];
    const float* mm2_b    = (const float*)d_in[18];
    const float* mm3_w    = (const float*)d_in[19];
    const float* mm3_b    = (const float*)d_in[20];
    const float* hg_w     = (const float*)d_in[21];
    const float* hg_b     = (const float*)d_in[22];
    const int*   hyper_nodes = (const int*)d_in[23];
    const int*   hyper_edges = (const int*)d_in[24];

    // ws layout: bfrag first (32B-aligned v16h array), then float arrays
    float* w = (float*)d_ws;
    v16h*  bfrag    = (v16h*)w;                       // 4*38*32 v16h = 38912 floats
    float* h1       = w + (size_t)Bsz * NCHUNK * 32 * 8;
    float* av1      = h1 + Bsz * Gn * Hd;
    float* bv1      = av1 + Bsz * Gn;
    float* upd1     = bv1 + Bsz * Gn;
    float* h2       = upd1 + Bsz * Gn * Hd;
    float* av2      = h2 + Bsz * Gn * Hd;
    float* bv2      = av2 + Bsz * Gn;
    float* upd2raw  = bv2 + Bsz * Gn;
    float* upd2     = upd2raw + Bsz * Gn * Hd;
    float* xf       = upd2 + Bsz * Gn * Hd;
    float* efeat    = xf + Bsz * Gn * Hd;

    const int nthread = 256;
    const int nbfrag  = Bsz * NCHUNK * 32;

    prep1_kernel<<<(Bsz * Gn + nthread - 1) / nthread, nthread, 0, stream>>>(
        x, infer_w, infer_b, mlp_e1_w, mlp_e1_b, h1, av1, bv1);

    bprep_kernel<<<(nbfrag + nthread - 1) / nthread, nthread, 0, stream>>>(h1, bfrag);

    round_kernel<<<Bsz * NTILE, 32, 0, stream>>>(
        h1, av1, bv1, edge1, bfrag, ALPHA, nodes1_w, nodes1_b, mm1_w, mm1_b, upd1);

    mid_kernel<<<(Gn + nthread - 1) / nthread, nthread, 0, stream>>>(
        upd1, W1, b1, mlp_e2_w, mlp_e2_b, h2, av2, bv2);

    bprep_kernel<<<(nbfrag + nthread - 1) / nthread, nthread, 0, stream>>>(h2, bfrag);

    round_kernel<<<Bsz * NTILE, 32, 0, stream>>>(
        h2, av2, bv2, edge2, bfrag, BETA, nodes2_w, nodes2_b, mm2_w, mm2_b, upd2raw);

    bn2_kernel<<<(Gn + nthread - 1) / nthread, nthread, 0, stream>>>(
        upd2raw, hg_w, upd2, xf);

    hyperedge_kernel<<<(NHE + nthread - 1) / nthread, nthread, 0, stream>>>(
        xf, hyper_nodes, hyper_edges, efeat);

    final_kernel<<<(Bsz * Gn + nthread - 1) / nthread, nthread, 0, stream>>>(
        upd2, efeat, hyper_nodes, hyper_edges, hg_b, mm3_w, mm3_b, (float*)d_out);
}
